// StandardCausalAttention_56367150793407
// MI455X (gfx1250) — matrix operand order repaired and checked by reading, not compiled, not run on hardware
//
#include <hip/hip_runtime.h>
#include <math.h>

// ---------------------------------------------------------------------------
// Causal MHA: x[2,2048,1024] -> out[2,2048,1024], 16 heads, head_dim 64.
// BF16 WMMA for all 4 matmul stages, f32 accum.
// GEMMs: double-buffered GLOBAL_LOAD_ASYNC_TO_LDS pipeline (ASYNCcnt).
// Flash attention: 32 q-rows/wave, 64-key tiles, exp2-domain online softmax,
// prefetched K/V streams, diagonal-only masking, lazy O-rescale.
// Workspace: ~50 MB.
// ---------------------------------------------------------------------------

typedef __bf16 bf16_t;
typedef __attribute__((ext_vector_type(16))) __bf16 v16bf;
typedef __attribute__((ext_vector_type(8)))  __bf16 v8bf;
typedef __attribute__((ext_vector_type(4)))  __bf16 v4bf;
typedef __attribute__((ext_vector_type(8)))  float  v8f;
typedef __attribute__((ext_vector_type(4)))  float  v4f;

#define EMBED   1024
#define NHEADS  16
#define HDIM    64
#define SEQ     2048
#define BATCH   2
#define BHEADS  (BATCH * NHEADS)   // 32
#define MROWS   (BATCH * SEQ)      // 4096
#define N3      (3 * EMBED)        // 3072

union ABu { v16bf v; v8bf h[2]; };

// A-matrix fragment (16 rows x 32 K): lane l (lo=l&15,hi=l>>4) row=lo,
// elems 0..7 = K[hi*8..+8], elems 8..15 = K[16+hi*8..+8].
static __device__ __forceinline__ v16bf load_frag_A(const bf16_t* base, int stride,
                                                    int lo, int hi) {
  ABu u;
  const bf16_t* p = base + lo * stride;
  u.h[0] = *(const v8bf*)(p + hi * 8);
  u.h[1] = *(const v8bf*)(p + 16 + hi * 8);
  return u.v;
}

// B-matrix fragment (32 K x 16 cols), source stored [n][k]: lane l col=lo,
// elems 0..7 = K[hi*8..+8], elems 8..15 = K[16+hi*8..+8].
static __device__ __forceinline__ v16bf load_frag_B(const bf16_t* base, int stride,
                                                    int lo, int hi) {
  ABu u;
  const bf16_t* p = base + lo * stride;
  u.h[0] = *(const v8bf*)(p + hi * 8);
  u.h[1] = *(const v8bf*)(p + 16 + hi * 8);
  return u.v;
}

static __device__ __forceinline__ v8f wmma_bf16(v16bf a, v16bf b, v8f c) {
  return __builtin_amdgcn_wmma_f32_16x16x32_bf16(false, a, false, b,
                                                 (short)0, c, false, false);
}

#define WAVE_LDS_SYNC()  asm volatile("s_wait_dscnt 0x0" ::: "memory")
#define ASYNC_WAIT0()    asm volatile("s_wait_asynccnt 0x0" ::: "memory")

// 16-byte global -> LDS async copy (DMA path, tracked by ASYNCcnt).
// Low 32 bits of a generic pointer to __shared__ are the LDS byte offset.
static __device__ __forceinline__ void async_copy16(bf16_t* lds_dst,
                                                    const bf16_t* gsrc) {
  unsigned lds_off = (unsigned)(unsigned long long)(uintptr_t)lds_dst;
  unsigned long long ga = (unsigned long long)(uintptr_t)gsrc;
  asm volatile("global_load_async_to_lds_b128 %0, %1, off"
               :: "v"(lds_off), "v"(ga) : "memory");
}

// ---------------------------------------------------------------------------
// Kernel 0a: fp32 [R][Ccol] -> bf16 transposed [Ccol][R]
// ---------------------------------------------------------------------------
__global__ __launch_bounds__(256) void transpose_to_bf16(const float* __restrict__ src,
                                                         bf16_t* __restrict__ dst,
                                                         int R, int Ccol) {
  size_t tid = (size_t)blockIdx.x * 256 + threadIdx.x;
  size_t total = (size_t)R * Ccol;
  if (tid >= total) return;
  int cc = (int)(tid / R);
  int r  = (int)(tid % R);
  dst[tid] = (bf16_t)src[(size_t)r * Ccol + cc];
}

// ---------------------------------------------------------------------------
// Kernel 0b: fp32 -> bf16 straight conversion (n % 4 == 0)
// ---------------------------------------------------------------------------
__global__ __launch_bounds__(256) void convert_to_bf16(const float* __restrict__ src,
                                                       bf16_t* __restrict__ dst,
                                                       int n4) {
  int t = blockIdx.x * 256 + threadIdx.x;
  if (t >= n4) return;
  v4f xv = *(const v4f*)(src + (size_t)t * 4);
  v4bf bv;
#pragma unroll
  for (int e = 0; e < 4; ++e) bv[e] = (bf16_t)xv[e];
  *(v4bf*)(dst + (size_t)t * 4) = bv;
}

// ---------------------------------------------------------------------------
// Unified 128x128 GEMM, K=1024, bf16 A [M][1024], bf16 B [N][1024] (pre-T).
// Double-buffered async-to-LDS, K-stage 64. 8 waves of 32x64 tiles.
// EPI=0: QKV epilogue (scatter Q*scale, K, V^T).  EPI=1: f32 out + bias.
// ---------------------------------------------------------------------------
template <int EPI>
__global__ __launch_bounds__(256) void gemm_bf16(const bf16_t* __restrict__ A,
                                                 const bf16_t* __restrict__ Bw,
                                                 const float* __restrict__ bias,
                                                 bf16_t* __restrict__ Qb,
                                                 bf16_t* __restrict__ Kb,
                                                 bf16_t* __restrict__ Vt,
                                                 float* __restrict__ Fo) {
  __shared__ __align__(16) bf16_t sA[2][128 * 64];   // 2 x 16 KB
  __shared__ __align__(16) bf16_t sB[2][128 * 64];   // 2 x 16 KB

  const int tid = threadIdx.x;
  const int l  = tid & 31;
  const int lo = l & 15;
  const int hi = l >> 4;
  const int w  = tid >> 5;       // wave 0..7
  const int wm = w >> 1;         // 0..3 -> 32-row strip
  const int wn = w & 1;          // 0..1 -> 64-col strip
  const int m0 = blockIdx.y * 128;
  const int n0 = blockIdx.x * 128;

  v8f acc[2][4];
#pragma unroll
  for (int i = 0; i < 2; ++i)
#pragma unroll
    for (int j = 0; j < 4; ++j) acc[i][j] = v8f{};

  // Each thread owns 4 16B chunks of the 128x64 A tile and 4 of B.
  auto issue_stage = [&](int buf, int tk) {
#pragma unroll
    for (int i = 0; i < 4; ++i) {
      int lin = i * 256 + tid;
      int r  = lin >> 3;          // 8 chunks per 64-elem row
      int kq = (lin & 7) << 3;
      async_copy16(&sA[buf][r * 64 + kq], A  + (size_t)(m0 + r) * EMBED + tk + kq);
      async_copy16(&sB[buf][r * 64 + kq], Bw + (size_t)(n0 + r) * EMBED + tk + kq);
    }
  };

  issue_stage(0, 0);
  ASYNC_WAIT0();
  __syncthreads();

  for (int s = 0; s < EMBED / 64; ++s) {
    const int cur = s & 1;
    if (s + 1 < EMBED / 64) issue_stage(cur ^ 1, (s + 1) * 64);

    v16bf af[2][2], bfr[4][2];
#pragma unroll
    for (int mf = 0; mf < 2; ++mf)
#pragma unroll
      for (int ks = 0; ks < 2; ++ks)
        af[mf][ks] = load_frag_A(&sA[cur][(wm * 32 + mf * 16) * 64 + ks * 32],
                                 64, lo, hi);
#pragma unroll
    for (int nf = 0; nf < 4; ++nf)
#pragma unroll
      for (int ks = 0; ks < 2; ++ks)
        bfr[nf][ks] = load_frag_B(&sB[cur][(wn * 64 + nf * 16) * 64 + ks * 32],
                                  64, lo, hi);
#pragma unroll
    for (int ks = 0; ks < 2; ++ks)
#pragma unroll
      for (int mf = 0; mf < 2; ++mf)
#pragma unroll
        for (int nf = 0; nf < 4; ++nf)
          acc[mf][nf] = wmma_bf16(af[mf][ks], bfr[nf][ks], acc[mf][nf]);

    ASYNC_WAIT0();        // own async copies for next stage have landed
    __syncthreads();      // everyone's have landed; prev-buffer reads done
  }

  // ---- epilogue ----
#pragma unroll
  for (int nf = 0; nf < 4; ++nf) {
    int gc = n0 + wn * 64 + nf * 16 + lo;
    float bv = bias[gc];
    if (EPI == 0) {
      int sec = gc >> 10;                 // 0=Q 1=K 2=V
      int c   = gc & 1023;
      int h   = c >> 6;
      int d   = c & 63;
#pragma unroll
      for (int mf = 0; mf < 2; ++mf)
#pragma unroll
        for (int g = 0; g < 8; ++g) {
          int gr = m0 + wm * 32 + mf * 16 + g + 8 * hi;
          int b  = gr >> 11;
          int t  = gr & 2047;
          int bh = b * NHEADS + h;
          float v = acc[mf][nf][g] + bv;
          if (sec == 0) {
            // fold 1/sqrt(64) * log2(e) into Q so scores land in exp2 domain
            Qb[((size_t)bh * SEQ + t) * HDIM + d] = (bf16_t)(v * 0.1803368802f);
          } else if (sec == 1) {
            Kb[((size_t)bh * SEQ + t) * HDIM + d] = (bf16_t)v;
          } else {
            Vt[((size_t)bh * HDIM + d) * SEQ + t] = (bf16_t)v;
          }
        }
    } else {
#pragma unroll
      for (int mf = 0; mf < 2; ++mf)
#pragma unroll
        for (int g = 0; g < 8; ++g) {
          int gr = m0 + wm * 32 + mf * 16 + g + 8 * hi;
          Fo[(size_t)gr * EMBED + gc] = acc[mf][nf][g] + bv;
        }
    }
  }
}

// ---------------------------------------------------------------------------
// Kernel 2: causal flash attention. One wave32 per (bh, 32-query-row tile),
// 64-key tiles. Scores arrive already in exp2 domain (scale folded into Q).
// ---------------------------------------------------------------------------
__global__ __launch_bounds__(32) void flash_attn(const bf16_t* __restrict__ Qb,
                                                 const bf16_t* __restrict__ Kb,
                                                 const bf16_t* __restrict__ Vt,
                                                 bf16_t* __restrict__ Ao) {
  __shared__ __align__(16) float  sS[32 * 64];   // 8 KB
  __shared__ __align__(16) bf16_t sP[32 * 64];   // 4 KB

  const int l  = threadIdx.x;
  const int lo = l & 15;
  const int hi = l >> 4;
  const int qt = blockIdx.x & (SEQ / 32 - 1);   // 0..63
  const int bh = blockIdx.x >> 6;               // 0..31
  const int q0 = qt * 32;

  const bf16_t* Qp = Qb + (size_t)bh * SEQ * HDIM;
  const bf16_t* Kp = Kb + (size_t)bh * SEQ * HDIM;
  const bf16_t* Vp = Vt + (size_t)bh * HDIM * SEQ;

  v16bf qa[2][2];
#pragma unroll
  for (int mf = 0; mf < 2; ++mf)
#pragma unroll
    for (int ks = 0; ks < 2; ++ks)
      qa[mf][ks] = load_frag_A(Qp + (size_t)(q0 + mf * 16) * HDIM + ks * 32,
                               HDIM, lo, hi);

  v8f o[2][4];
#pragma unroll
  for (int i = 0; i < 2; ++i)
#pragma unroll
    for (int j = 0; j < 4; ++j) o[i][j] = v8f{};

  float mrow = -__builtin_inff();
  float lrow = 0.0f;

  const int ntiles = (q0 + 32 + 63) >> 6;   // 64-key tiles covering 0..q0+31
  for (int kt = 0; kt < ntiles; ++kt) {
    const int k0 = kt * 64;

    // ---- prefetch V tile (used after softmax) and next K tile ----
#pragma unroll
    for (int nf = 0; nf < 4; ++nf) {
      __builtin_prefetch(Vp + (size_t)(nf * 16 + lo) * SEQ + k0 + hi * 16, 0, 3);
      __builtin_prefetch(Vp + (size_t)(nf * 16 + lo) * SEQ + k0 + 32 + hi * 16, 0, 3);
    }
    if (kt + 1 < ntiles) {
#pragma unroll
      for (int nf = 0; nf < 4; ++nf)
        __builtin_prefetch(Kp + (size_t)(k0 + 64 + nf * 16 + lo) * HDIM + hi * 16,
                           0, 3);
    }

    // ---- S = Q K^T (32x64 tile, exp2-domain, f32 accum) ----
    v8f s[2][4];
#pragma unroll
    for (int i = 0; i < 2; ++i)
#pragma unroll
      for (int j = 0; j < 4; ++j) s[i][j] = v8f{};
#pragma unroll
    for (int ks = 0; ks < 2; ++ks) {
      v16bf kb[4];                         // 4 loads in flight per k-step
#pragma unroll
      for (int nf = 0; nf < 4; ++nf)
        kb[nf] = load_frag_B(Kp + (size_t)(k0 + nf * 16) * HDIM + ks * 32,
                             HDIM, lo, hi);
#pragma unroll
      for (int mf = 0; mf < 2; ++mf)
#pragma unroll
        for (int nf = 0; nf < 4; ++nf)
          s[mf][nf] = wmma_bf16(qa[mf][ks], kb[nf], s[mf][nf]);
    }

    // ---- causal mask (diagonal tiles only) + C/D layout -> LDS ----
    const bool needmask = (k0 + 63 > q0);
#pragma unroll
    for (int mf = 0; mf < 2; ++mf)
#pragma unroll
      for (int nf = 0; nf < 4; ++nf)
#pragma unroll
        for (int g = 0; g < 8; ++g) {
          int rloc = mf * 16 + g + 8 * hi;
          int cloc = nf * 16 + lo;
          float v = s[mf][nf][g];
          if (needmask && (k0 + cloc > q0 + rloc)) v = -1.0e30f;
          sS[rloc * 64 + cloc] = v;
        }
    WAVE_LDS_SYNC();

    // ---- online softmax: lane owns query row r = l; two LDS sweeps ----
    float smax = -__builtin_inff();
#pragma unroll
    for (int j4 = 0; j4 < 16; ++j4) {
      v4f rd = *(const v4f*)(sS + l * 64 + j4 * 4);
#pragma unroll
      for (int e = 0; e < 4; ++e) smax = fmaxf(smax, rd[e]);
    }
    float mnew  = fmaxf(mrow, smax);
    float alpha = exp2f(mrow - mnew);       // 0 on first tile
    float psum  = 0.0f;
#pragma unroll
    for (int j4 = 0; j4 < 16; ++j4) {
      v4f rd = *(const v4f*)(sS + l * 64 + j4 * 4);
      v4bf pb;
#pragma unroll
      for (int e = 0; e < 4; ++e) {
        float p = exp2f(rd[e] - mnew);
        psum += p;
        pb[e] = (bf16_t)p;
      }
      *(v4bf*)(sP + l * 64 + j4 * 4) = pb;
    }
    lrow = lrow * alpha + psum;

    // ---- rescale O only if some row max actually moved ----
    if (!__all(smax <= mrow)) {
#pragma unroll
      for (int mf = 0; mf < 2; ++mf)
#pragma unroll
        for (int g = 0; g < 8; ++g) {
          float af = __shfl(alpha, mf * 16 + g + 8 * hi, 32);
#pragma unroll
          for (int nf = 0; nf < 4; ++nf) o[mf][nf][g] *= af;
        }
    }
    mrow = mnew;
    WAVE_LDS_SYNC();

    // ---- O += P V over two 32-key chunks ----
#pragma unroll
    for (int ks2 = 0; ks2 < 2; ++ks2) {
      v16bf pa[2], vb[4];
#pragma unroll
      for (int nf = 0; nf < 4; ++nf)
        vb[nf] = load_frag_B(Vp + (size_t)(nf * 16) * SEQ + k0 + ks2 * 32,
                             SEQ, lo, hi);
#pragma unroll
      for (int mf = 0; mf < 2; ++mf)
        pa[mf] = load_frag_A(sP + (mf * 16) * 64 + ks2 * 32, 64, lo, hi);
#pragma unroll
      for (int mf = 0; mf < 2; ++mf)
#pragma unroll
        for (int nf = 0; nf < 4; ++nf)
          o[mf][nf] = wmma_bf16(pa[mf], vb[nf], o[mf][nf]);
    }
    WAVE_LDS_SYNC();
  }

  // ---- normalize and write merged-head bf16 output [B*T][C] ----
  float linv = 1.0f / lrow;
  const int b = bh >> 4;
  const int h = bh & 15;
#pragma unroll
  for (int mf = 0; mf < 2; ++mf)
#pragma unroll
    for (int g = 0; g < 8; ++g) {
      int t = q0 + mf * 16 + g + 8 * hi;
      float li = __shfl(linv, mf * 16 + g + 8 * hi, 32);
      size_t rowbase = ((size_t)b * SEQ + t) * EMBED + h * HDIM;
#pragma unroll
      for (int nf = 0; nf < 4; ++nf)
        Ao[rowbase + nf * 16 + lo] = (bf16_t)(o[mf][nf][g] * li);
    }
}

// ---------------------------------------------------------------------------
// Host-side orchestration
// ---------------------------------------------------------------------------
extern "C" void kernel_launch(void* const* d_in, const int* in_sizes, int n_in,
                              void* d_out, int out_size, void* d_ws, size_t ws_size,
                              hipStream_t stream) {
  const float* x    = (const float*)d_in[0];   // [2,2048,1024]
  const float* Wqkv = (const float*)d_in[1];   // [1024,3072]
  const float* bqkv = (const float*)d_in[2];   // [3072]
  const float* Wout = (const float*)d_in[3];   // [1024,1024]
  const float* bout = (const float*)d_in[4];   // [1024]
  float* out = (float*)d_out;                  // [2,2048,1024]

  char* ws = (char*)d_ws;
  size_t off = 0;
  auto alloc = [&](size_t bytes) {
    char* p = ws + off;
    off += (bytes + 255) & ~(size_t)255;
    return p;
  };
  bf16_t* Xb    = (bf16_t*)alloc((size_t)MROWS * EMBED * 2);        // 8 MB
  bf16_t* WqkvT = (bf16_t*)alloc((size_t)N3 * EMBED * 2);           // 6 MB
  bf16_t* WoutT = (bf16_t*)alloc((size_t)EMBED * EMBED * 2);        // 2 MB
  bf16_t* Qb    = (bf16_t*)alloc((size_t)BHEADS * SEQ * HDIM * 2);  // 8 MB
  bf16_t* Kb    = (bf16_t*)alloc((size_t)BHEADS * SEQ * HDIM * 2);  // 8 MB
  bf16_t* Vt    = (bf16_t*)alloc((size_t)BHEADS * HDIM * SEQ * 2);  // 8 MB
  bf16_t* Ao    = (bf16_t*)alloc((size_t)MROWS * EMBED * 2);        // 8 MB
  (void)ws_size;

  // 0) precision/layout prep
  convert_to_bf16<<<(MROWS * EMBED / 4 + 255) / 256, 256, 0, stream>>>(
      x, Xb, MROWS * EMBED / 4);
  transpose_to_bf16<<<(EMBED * N3 + 255) / 256, 256, 0, stream>>>(
      Wqkv, WqkvT, EMBED, N3);
  transpose_to_bf16<<<(EMBED * EMBED + 255) / 256, 256, 0, stream>>>(
      Wout, WoutT, EMBED, EMBED);
  // 1) QKV projection (async-pipelined GEMM)
  gemm_bf16<0><<<dim3(N3 / 128, MROWS / 128), 256, 0, stream>>>(
      Xb, WqkvT, bqkv, Qb, Kb, Vt, nullptr);
  // 2) causal flash attention
  flash_attn<<<dim3(BHEADS * (SEQ / 32)), 32, 0, stream>>>(Qb, Kb, Vt, Ao);
  // 3) output projection
  gemm_bf16<1><<<dim3(EMBED / 128, MROWS / 128), 256, 0, stream>>>(
      Ao, WoutT, bout, nullptr, nullptr, nullptr, out);
}
